// Supervised_NTXentLoss_4964982194342
// MI455X (gfx1250) — matrix operand order repaired and checked by reading, not compiled, not run on hardware
//
#include <hip/hip_runtime.h>
#include <hip/hip_bf16.h>
#include <math.h>
#include <stdint.h>

// ---------------------------------------------------------------------------
// Supervised NT-Xent loss, fused for MI455X (gfx1250).
//   N=8192, D=512, T=0.5. S = norm(out0) @ norm(out1)^T / T, fused exp +
//   masked row sums; 68.7 GFLOP matmul, f16 operands (16 MB) live in L2.
// Matrix path: v_wmma_f32_16x16x32_f16, A panel register-resident,
// B panel staged to LDS once per block via global_load_async_to_lds_b128
// (ASYNCcnt), consumed with ds_load_b128 software-pipelined 2 steps ahead.
// ---------------------------------------------------------------------------

typedef __attribute__((ext_vector_type(16))) _Float16 v16h;
typedef __attribute__((ext_vector_type(8)))  _Float16 v8h;
typedef __attribute__((ext_vector_type(8)))  float    v8f;

#define NROWS 8192
#define NDIM  512
#define INV_T 2.0f
#define COL_BLOCKS 8
#define COLS_PER_BLOCK (NROWS / COL_BLOCKS)   // 1024
#define TILES_PER_BLOCK (COLS_PER_BLOCK / 16) // 64
#define ROWS_PER_BLOCK 128                    // 8 waves * 16 rows
#define ROW_BLOCKS (NROWS / ROWS_PER_BLOCK)   // 64

// B panel in LDS: 16 columns x 520 halves (512 + 8 pad halves -> column
// stride 1040 B = 260 dwords == 4 mod 64 banks: conflict-free b128 reads).
#define BCOL_HALVES 520
#define BUF_HALVES  (16 * BCOL_HALVES)        // 8320 halves
#define BUF_BYTES   (BUF_HALVES * 2)          // 16640 bytes

__device__ __forceinline__ v16h cat16(v8h lo, v8h hi) {
    return __builtin_shufflevector(lo, hi, 0,1,2,3,4,5,6,7,8,9,10,11,12,13,14,15);
}

// ---------------------------------------------------------------------------
// Kernel 1: L2-normalize rows and convert to f16. One wave per row.
// ---------------------------------------------------------------------------
__global__ __launch_bounds__(256) void normalize_to_f16(
    const float* __restrict__ src, _Float16* __restrict__ dst)
{
    const int wave = threadIdx.x >> 5;
    const int lane = threadIdx.x & 31;
    const int row  = blockIdx.x * 8 + wave;

    const float4* p = (const float4*)(src + (size_t)row * NDIM + lane * 16);
    float4 x0 = p[0], x1 = p[1], x2 = p[2], x3 = p[3];

    float s = x0.x*x0.x + x0.y*x0.y + x0.z*x0.z + x0.w*x0.w
            + x1.x*x1.x + x1.y*x1.y + x1.z*x1.z + x1.w*x1.w
            + x2.x*x2.x + x2.y*x2.y + x2.z*x2.z + x2.w*x2.w
            + x3.x*x3.x + x3.y*x3.y + x3.z*x3.z + x3.w*x3.w;
    #pragma unroll
    for (int off = 16; off >= 1; off >>= 1) s += __shfl_xor(s, off, 32);

    const float inv = 1.0f / fmaxf(sqrtf(s), 1e-12f);

    v8h h0, h1;
    h0[0]=(_Float16)(x0.x*inv); h0[1]=(_Float16)(x0.y*inv);
    h0[2]=(_Float16)(x0.z*inv); h0[3]=(_Float16)(x0.w*inv);
    h0[4]=(_Float16)(x1.x*inv); h0[5]=(_Float16)(x1.y*inv);
    h0[6]=(_Float16)(x1.z*inv); h0[7]=(_Float16)(x1.w*inv);
    h1[0]=(_Float16)(x2.x*inv); h1[1]=(_Float16)(x2.y*inv);
    h1[2]=(_Float16)(x2.z*inv); h1[3]=(_Float16)(x2.w*inv);
    h1[4]=(_Float16)(x3.x*inv); h1[5]=(_Float16)(x3.y*inv);
    h1[6]=(_Float16)(x3.z*inv); h1[7]=(_Float16)(x3.w*inv);

    v8h* q = (v8h*)(dst + (size_t)row * NDIM + lane * 16);
    q[0] = h0; q[1] = h1;
}

// ---------------------------------------------------------------------------
// Kernel 2: fused WMMA GEMM + exp + masked row sums.
// 8 waves/block, wave w owns 16 rows (A panel hoisted to 128 VGPRs).
// B column panel staged to LDS (async, double buffered); B fragments read
// via ds_load_b128 with an explicit 3-buffer rotation issued 2 steps ahead.
// ---------------------------------------------------------------------------
__global__ __launch_bounds__(256, 1) void ntxent_wmma(
    const _Float16* __restrict__ Ah,
    const _Float16* __restrict__ Bh,
    const int*      __restrict__ labels,
    float* __restrict__ posPart,       // [COL_BLOCKS][NROWS]
    float* __restrict__ allPart)       // [COL_BLOCKS][NROWS]
{
    __shared__ __align__(16) _Float16 Bs[2][BUF_HALVES];

    const int wave = threadIdx.x >> 5;
    const int lane = threadIdx.x & 31;
    const int rowWaveBase = blockIdx.x * ROWS_PER_BLOCK + wave * 16;
    const int colBlock = blockIdx.y;

    const uint64_t bbase64 = (uint64_t)(uintptr_t)Bh;
    const uint32_t ldsBase = (uint32_t)(uintptr_t)&Bs[0][0];

    // ---- A fragments: lane = row (lane&15); K-halves split by lane>=16 ----
    const int arow  = rowWaveBase + (lane & 15);
    const int ahalf = (lane >> 4) * 16;
    const _Float16* abase = Ah + (size_t)arow * NDIM;
    v16h afrag[16];
    #pragma unroll
    for (int kk = 0; kk < 16; ++kk) {
        v8h lo = *(const v8h*)(abase + kk * 32 + ahalf);
        v8h hi = *(const v8h*)(abase + kk * 32 + 8 + ahalf);
        afrag[kk] = cat16(lo, hi);
    }

    // ---- row labels for this lane's 8 output rows ----
    int li[8];
    #pragma unroll
    for (int r = 0; r < 8; ++r)
        li[r] = labels[rowWaveBase + r + 8 * (lane >> 4)];

    float accAll[8], accPos[8];
    #pragma unroll
    for (int r = 0; r < 8; ++r) { accAll[r] = 0.0f; accPos[r] = 0.0f; }

    const int koff = (lane >> 4) * 16;   // B K-half select per lane group

    // ---- async stage of one 16-column B panel into LDS buffer `buf` ----
    // 1024 16-byte chunks, 4 per thread. GVS addressing: SGPR64 base +
    // signed 32-bit VGPR byte offset; VDST = LDS byte address.
    auto stage = [&](int ct, int buf) {
        const uint32_t tileBase =
            (uint32_t)(colBlock * COLS_PER_BLOCK + ct * 16) * (NDIM * 2);
        #pragma unroll
        for (int i = 0; i < 4; ++i) {
            const int chunk = (int)threadIdx.x + i * 256;   // 0..1023
            const int col = chunk >> 6;                     // 16 columns
            const int kc  = chunk & 63;                     // 64 chunks/col
            const uint32_t goff = tileBase + (uint32_t)(col * (NDIM * 2) + kc * 16);
            const uint32_t loff = ldsBase +
                (uint32_t)(buf * BUF_BYTES + col * (BCOL_HALVES * 2) + kc * 16);
            asm volatile("global_load_async_to_lds_b128 %0, %1, %2"
                         :: "v"(loff), "v"(goff), "s"(bbase64)
                         : "memory");
        }
    };

    stage(0, 0);
    int buf = 0;

    for (int ct = 0; ct < TILES_PER_BLOCK; ++ct) {
        // stage(ct) complete (own wave) + all waves past previous compute
        asm volatile("s_wait_asynccnt 0x0" ::: "memory");
        __syncthreads();

        if (ct + 1 < TILES_PER_BLOCK) stage(ct + 1, buf ^ 1);

        const int colBase = colBlock * COLS_PER_BLOCK + ct * 16;
        const int col = colBase + (lane & 15);
        const _Float16* bcol = &Bs[buf][(lane & 15) * BCOL_HALVES];

        // ---- WMMA chain with B fragments pipelined 2 k-steps ahead ----
        v8h lo[3], hi[3];
        lo[0] = *(const v8h*)(bcol + koff);
        hi[0] = *(const v8h*)(bcol + koff + 8);
        lo[1] = *(const v8h*)(bcol + 32 + koff);
        hi[1] = *(const v8h*)(bcol + 32 + koff + 8);

        v8f c = {};
        #pragma unroll
        for (int kk = 0; kk < 16; ++kk) {
            const int cur = kk % 3;
            const int nxt = (kk + 2) % 3;
            if (kk + 2 < 16) {
                lo[nxt] = *(const v8h*)(bcol + (kk + 2) * 32 + koff);
                hi[nxt] = *(const v8h*)(bcol + (kk + 2) * 32 + koff + 8);
            }
            c = __builtin_amdgcn_wmma_f32_16x16x32_f16(
                    false, afrag[kk], false, cat16(lo[cur], hi[cur]),
                    (short)0, c, false, false);
        }

        // ---- epilogue: scale, exp, mask, accumulate ----
        const int lj = labels[col];
        #pragma unroll
        for (int r = 0; r < 8; ++r) {
            float e = __expf(c[r] * INV_T);
            accAll[r] += e;
            if (li[r] == lj) accPos[r] += e;
        }
        buf ^= 1;
    }

    // ---- cross-lane (16-wide segment) reduction: sum over columns ----
    #pragma unroll
    for (int r = 0; r < 8; ++r) {
        float a = accAll[r], p = accPos[r];
        #pragma unroll
        for (int off = 8; off >= 1; off >>= 1) {
            a += __shfl_xor(a, off, 16);
            p += __shfl_xor(p, off, 16);
        }
        if ((lane & 15) == 0) {
            const int row = rowWaveBase + r + 8 * (lane >> 4);
            allPart[colBlock * NROWS + row] = a;
            posPart[colBlock * NROWS + row] = p;
        }
    }
}

// ---------------------------------------------------------------------------
// Kernel 3: per-row log(pos/all) from the 8 col-block partials.
// ---------------------------------------------------------------------------
__global__ __launch_bounds__(256) void row_log(
    const float* __restrict__ posPart, const float* __restrict__ allPart,
    float* __restrict__ rowLoss)
{
    const int i = blockIdx.x * 256 + threadIdx.x;
    float p = 0.0f, a = 0.0f;
    #pragma unroll
    for (int b = 0; b < COL_BLOCKS; ++b) {
        p += posPart[b * NROWS + i];
        a += allPart[b * NROWS + i];
    }
    rowLoss[i] = __logf(p / a);
}

// ---------------------------------------------------------------------------
// Kernel 4: single-block LDS tree reduction -> scalar loss.
// ---------------------------------------------------------------------------
__global__ __launch_bounds__(256) void final_reduce(
    const float* __restrict__ rowLoss, float* __restrict__ out)
{
    __shared__ float sm[256];
    float s = 0.0f;
    for (int i = threadIdx.x; i < NROWS; i += 256) s += rowLoss[i];
    sm[threadIdx.x] = s;
    __syncthreads();
    #pragma unroll
    for (int off = 128; off > 0; off >>= 1) {
        if ((int)threadIdx.x < off) sm[threadIdx.x] += sm[threadIdx.x + off];
        __syncthreads();
    }
    if (threadIdx.x == 0) out[0] = -sm[0] / (float)NROWS;
}

// ---------------------------------------------------------------------------
// Host launcher. Workspace layout (~16.6 MB):
//   [0, 8MB)    Ah f16    [8MB, 16MB) Bh f16
//   [16MB, ..)  allPart[8][8192], posPart[8][8192], rowLoss[8192] (f32)
// ---------------------------------------------------------------------------
extern "C" void kernel_launch(void* const* d_in, const int* in_sizes, int n_in,
                              void* d_out, int out_size, void* d_ws, size_t ws_size,
                              hipStream_t stream)
{
    const float* out0   = (const float*)d_in[0];
    const float* out1   = (const float*)d_in[1];
    const int*   labels = (const int*)d_in[2];

    char* ws = (char*)d_ws;
    _Float16* Ah = (_Float16*)ws;
    _Float16* Bh = (_Float16*)(ws + ((size_t)8 << 20));
    float* allPart = (float*)(ws + ((size_t)16 << 20));
    float* posPart = allPart + COL_BLOCKS * NROWS;
    float* rowLoss = posPart + COL_BLOCKS * NROWS;

    normalize_to_f16<<<NROWS / 8, 256, 0, stream>>>(out0, Ah);
    normalize_to_f16<<<NROWS / 8, 256, 0, stream>>>(out1, Bh);

    dim3 grid(ROW_BLOCKS, COL_BLOCKS, 1);
    ntxent_wmma<<<grid, 256, 0, stream>>>(Ah, Bh, labels, posPart, allPart);

    row_log<<<NROWS / 256, 256, 0, stream>>>(posPart, allPart, rowLoss);
    final_reduce<<<1, 256, 0, stream>>>(rowLoss, (float*)d_out);
}
